// GraphSAGE_32830730011134
// MI455X (gfx1250) — matrix operand order repaired and checked by reading, not compiled, not run on hardware
//
#include <hip/hip_runtime.h>
#include <hip/hip_bf16.h>

#ifndef __has_builtin
#define __has_builtin(x) 0
#endif

#if __has_builtin(__builtin_amdgcn_global_load_async_to_lds_b128)
#define GIN_ASYNC_LDS 1
#else
#define GIN_ASYNC_LDS 0
#endif

// ---------------- problem constants ----------------
#define N_NODES  10000
#define N_EDGES  160000
#define F_IN     128
#define HIDDEN   512
#define N_CLASSES 10
#define N_GRAPHS 64

// ---------------- WMMA types ----------------
typedef __bf16 bf16_t;
typedef bf16_t v16bf __attribute__((ext_vector_type(16)));
typedef float  v8f   __attribute__((ext_vector_type(8)));
typedef int    gin_v4i __attribute__((vector_size(16)));

union FragBF {
    uint4 u[2];   // 32 bytes
    v16bf v;
};

__device__ __forceinline__ unsigned short f32_to_bf16(float f) {
    unsigned int u = __float_as_uint(f);
    u += 0x7fffu + ((u >> 16) & 1u);   // round-to-nearest-even
    return (unsigned short)(u >> 16);
}

__device__ __forceinline__ void gin_async_cp16(const unsigned short* g, unsigned short* l) {
#if GIN_ASYNC_LDS
    __builtin_amdgcn_global_load_async_to_lds_b128(
        (__attribute__((address_space(1))) gin_v4i*)(gin_v4i*)(void*)const_cast<unsigned short*>(g),
        (__attribute__((address_space(3))) gin_v4i*)(gin_v4i*)(void*)l,
        0, 0);
#else
    (void)g; (void)l;
#endif
}

__device__ __forceinline__ void gin_wait_async() {
#if GIN_ASYNC_LDS
#if __has_builtin(__builtin_amdgcn_s_wait_asynccnt)
    __builtin_amdgcn_s_wait_asynccnt(0);
#else
    asm volatile("s_wait_asynccnt 0x0" ::: "memory");
#endif
#endif
}

// ---------------- elementwise kernels ----------------
__global__ void gin_fill_zero(float* __restrict__ p, int n) {
    int i = blockIdx.x * blockDim.x + threadIdx.x;
    if (i < n) p[i] = 0.0f;
}

// W: [K][N] f32 row-major  ->  Wt: [N][K] bf16
__global__ void gin_wt_convert(const float* __restrict__ W, unsigned short* __restrict__ Wt,
                               int K, int N) {
    int idx = blockIdx.x * blockDim.x + threadIdx.x;
    if (idx >= K * N) return;
    int k = idx / N, n = idx % N;
    Wt[(size_t)n * K + k] = f32_to_bf16(W[idx]);
}

// aggr[dst[e]][f..f+3] += h[src[e]][f..f+3]   (thread per (edge, float4 group))
__global__ void gin_aggregate(const float4* __restrict__ h4,
                              const int* __restrict__ src, const int* __restrict__ dst,
                              float* __restrict__ aggr, int logF4, int total) {
    int i = blockIdx.x * blockDim.x + threadIdx.x;
    if (i >= total) return;
    int e = i >> logF4;
    int g = i & ((1 << logF4) - 1);
    float4 v = h4[((size_t)src[e] << logF4) + g];
    float* b = aggr + (((size_t)dst[e] << logF4) + g) * 4;
    atomicAdd(b + 0, v.x);
    atomicAdd(b + 1, v.y);
    atomicAdd(b + 2, v.z);
    atomicAdd(b + 3, v.w);
}

// z_bf16 = bf16(h + aggr)
__global__ void gin_add_convert(const float* __restrict__ h, const float* __restrict__ a,
                                unsigned short* __restrict__ z, int n) {
    int i = blockIdx.x * blockDim.x + threadIdx.x;
    if (i < n) z[i] = f32_to_bf16(h[i] + a[i]);
}

// ---------------- tiled WMMA GEMM:  C = relu(A @ Wt^T + bias) ----------------
// A:  [M][K]  bf16 row-major
// Wt: [N][K]  bf16 row-major (weight pre-transposed)
// C:  [M][N]  bf16 (OUT_BF16) or f32
// Block: 256 thr = 8 waves in 4(M) x 2(N); wave tile 32x64; block tile 128x128.
// Double-buffered LDS; async global->LDS when toolchain provides the builtin.
#define BM 128
#define BN 128
#define BK 32
#define LDA 40   // halfword stride (80B: 16B aligned + bank spreading)
#define LDB 40

template <bool OUT_BF16>
__global__ __launch_bounds__(256)
void gin_gemm_bias_relu(const unsigned short* __restrict__ A,
                        const unsigned short* __restrict__ Wt,
                        const float* __restrict__ bias,
                        void* __restrict__ Cout,
                        int M, int K, int N) {
    __shared__ __align__(16) unsigned short sA[2][BM * LDA];
    __shared__ __align__(16) unsigned short sB[2][BN * LDB];

    const int tid  = threadIdx.x;
    const int wave = tid >> 5;
    const int lane = tid & 31;

    const int bm = blockIdx.x * BM;
    const int bn = blockIdx.y * BN;
    const int m0     = bm + (wave >> 1) * 32;   // wave rows m0..m0+31
    const int n0base = (wave & 1) * 64;         // wave cols (in-tile)

    // staging geometry: A tile = 512 uint4, B tile = 512 uint4; 2+2 per thread
    const int srow = tid >> 2;                  // 0..63  (also srow+64)
    const int schk = (tid & 3) * 8;             // element offset of 8-bf16 chunk

    // fragment geometry (ISA 7.12.2, wave32)
    const int frow  = lane & 15;
    const int akoff = (lane >> 4) * 8;
    const int bkoff = (lane >> 4) * 8;

    v8f zero;
    #pragma unroll
    for (int j = 0; j < 8; ++j) zero[j] = 0.0f;
    v8f acc[2][4];
    #pragma unroll
    for (int s = 0; s < 2; ++s)
        #pragma unroll
        for (int i = 0; i < 4; ++i) acc[s][i] = zero;

    const uint4 z4 = make_uint4(0u, 0u, 0u, 0u);
    const int nsteps = K / BK;

    // ---------------- prologue: stage tile 0 into buffer 0 ----------------
    {
        const int k0 = 0;
        int g0 = bm + srow, g1 = g0 + 64;
#if GIN_ASYNC_LDS
        if (g0 < M) gin_async_cp16(A + (size_t)g0 * K + k0 + schk, &sA[0][srow * LDA + schk]);
        else        *(uint4*)&sA[0][srow * LDA + schk] = z4;
        if (g1 < M) gin_async_cp16(A + (size_t)g1 * K + k0 + schk, &sA[0][(srow + 64) * LDA + schk]);
        else        *(uint4*)&sA[0][(srow + 64) * LDA + schk] = z4;
        gin_async_cp16(Wt + (size_t)(bn + srow) * K + k0 + schk,      &sB[0][srow * LDB + schk]);
        gin_async_cp16(Wt + (size_t)(bn + srow + 64) * K + k0 + schk, &sB[0][(srow + 64) * LDB + schk]);
        gin_wait_async();
#else
        uint4 ra0 = (g0 < M) ? *(const uint4*)(A + (size_t)g0 * K + k0 + schk) : z4;
        uint4 ra1 = (g1 < M) ? *(const uint4*)(A + (size_t)g1 * K + k0 + schk) : z4;
        uint4 rb0 = *(const uint4*)(Wt + (size_t)(bn + srow) * K + k0 + schk);
        uint4 rb1 = *(const uint4*)(Wt + (size_t)(bn + srow + 64) * K + k0 + schk);
        *(uint4*)&sA[0][srow * LDA + schk]        = ra0;
        *(uint4*)&sA[0][(srow + 64) * LDA + schk] = ra1;
        *(uint4*)&sB[0][srow * LDB + schk]        = rb0;
        *(uint4*)&sB[0][(srow + 64) * LDB + schk] = rb1;
#endif
    }
    __syncthreads();

    // ---------------- main loop (double buffered) ----------------
    int buf = 0;
    for (int s = 0; s < nsteps; ++s) {
        const bool hasNext = (s + 1) < nsteps;
        const int kn = (s + 1) * BK;
        const int nb = buf ^ 1;
        int g0 = bm + srow, g1 = g0 + 64;

#if GIN_ASYNC_LDS
        if (hasNext) {
            if (g0 < M) gin_async_cp16(A + (size_t)g0 * K + kn + schk, &sA[nb][srow * LDA + schk]);
            else        *(uint4*)&sA[nb][srow * LDA + schk] = z4;
            if (g1 < M) gin_async_cp16(A + (size_t)g1 * K + kn + schk, &sA[nb][(srow + 64) * LDA + schk]);
            else        *(uint4*)&sA[nb][(srow + 64) * LDA + schk] = z4;
            gin_async_cp16(Wt + (size_t)(bn + srow) * K + kn + schk,      &sB[nb][srow * LDB + schk]);
            gin_async_cp16(Wt + (size_t)(bn + srow + 64) * K + kn + schk, &sB[nb][(srow + 64) * LDB + schk]);
        }
#else
        uint4 ra0 = z4, ra1 = z4, rb0 = z4, rb1 = z4;
        if (hasNext) {
            if (g0 < M) ra0 = *(const uint4*)(A + (size_t)g0 * K + kn + schk);
            if (g1 < M) ra1 = *(const uint4*)(A + (size_t)g1 * K + kn + schk);
            rb0 = *(const uint4*)(Wt + (size_t)(bn + srow) * K + kn + schk);
            rb1 = *(const uint4*)(Wt + (size_t)(bn + srow + 64) * K + kn + schk);
        }
#endif

        // ---- compute from current buffer: 2 A-frags x 4 B-frags = 8 WMMA ----
        {
            FragBF a0, a1;
            const unsigned short* pa0 = &sA[buf][((wave >> 1) * 32 + frow) * LDA];
            const unsigned short* pa1 = pa0 + 16 * LDA;
            a0.u[0] = *(const uint4*)(pa0 + akoff);
            a0.u[1] = *(const uint4*)(pa0 + akoff + 16);
            a1.u[0] = *(const uint4*)(pa1 + akoff);
            a1.u[1] = *(const uint4*)(pa1 + akoff + 16);
            #pragma unroll
            for (int ns = 0; ns < 4; ++ns) {
                FragBF b;
                const unsigned short* pb = &sB[buf][(n0base + ns * 16 + frow) * LDB + bkoff];
                b.u[0] = *(const uint4*)(pb);
                b.u[1] = *(const uint4*)(pb + 16);
                acc[0][ns] = __builtin_amdgcn_wmma_f32_16x16x32_bf16(
                    false, a0.v, false, b.v, (short)0, acc[0][ns], false, false);
                acc[1][ns] = __builtin_amdgcn_wmma_f32_16x16x32_bf16(
                    false, a1.v, false, b.v, (short)0, acc[1][ns], false, false);
            }
        }

        if (hasNext) {
#if GIN_ASYNC_LDS
            gin_wait_async();
#else
            *(uint4*)&sA[nb][srow * LDA + schk]        = ra0;
            *(uint4*)&sA[nb][(srow + 64) * LDA + schk] = ra1;
            *(uint4*)&sB[nb][srow * LDB + schk]        = rb0;
            *(uint4*)&sB[nb][(srow + 64) * LDB + schk] = rb1;
#endif
        }
        __syncthreads();
        buf ^= 1;
    }

    // ---------------- epilogue: bias + relu, store ----------------
    const int crow0 = m0 + ((lane >> 4) * 8);
    const int ccol0 = bn + n0base + (lane & 15);
    #pragma unroll
    for (int sub = 0; sub < 2; ++sub) {
        #pragma unroll
        for (int ns = 0; ns < 4; ++ns) {
            int col = ccol0 + ns * 16;
            float bv = bias[col];
            #pragma unroll
            for (int r = 0; r < 8; ++r) {
                int row = crow0 + sub * 16 + r;
                if (row < M) {
                    float v = acc[sub][ns][r] + bv;
                    v = v > 0.0f ? v : 0.0f;
                    if (OUT_BF16)
                        ((unsigned short*)Cout)[(size_t)row * N + col] = f32_to_bf16(v);
                    else
                        ((float*)Cout)[(size_t)row * N + col] = v;
                }
            }
        }
    }
}

// ---------------- pooling + classifier ----------------
__global__ void gin_pool_sum(const float4* __restrict__ h4, const int* __restrict__ batch,
                             float* __restrict__ sums, int total) {
    int i = blockIdx.x * blockDim.x + threadIdx.x;
    if (i >= total) return;
    int node = i >> 7;          // HIDDEN/4 = 128 groups per node
    int g    = i & 127;
    float4 v = h4[i];
    float* b = sums + ((size_t)batch[node] * HIDDEN) + g * 4;
    atomicAdd(b + 0, v.x);
    atomicAdd(b + 1, v.y);
    atomicAdd(b + 2, v.z);
    atomicAdd(b + 3, v.w);
}

__global__ void gin_count_nodes(const int* __restrict__ batch, float* __restrict__ counts) {
    int i = blockIdx.x * blockDim.x + threadIdx.x;
    if (i < N_NODES) atomicAdd(&counts[batch[i]], 1.0f);
}

__global__ void gin_fc_logsoftmax(const float* __restrict__ sums, const float* __restrict__ counts,
                                  const float* __restrict__ fc_w, const float* __restrict__ fc_b,
                                  float* __restrict__ out) {
    __shared__ float logits[N_GRAPHS * N_CLASSES];
    for (int p = threadIdx.x; p < N_GRAPHS * N_CLASSES; p += blockDim.x) {
        int g = p / N_CLASSES, c = p % N_CLASSES;
        float inv = 1.0f / fmaxf(counts[g], 1.0f);
        float acc = fc_b[c];
        for (int k = 0; k < HIDDEN; ++k)
            acc += sums[(size_t)g * HIDDEN + k] * inv * fc_w[k * N_CLASSES + c];
        logits[p] = acc;
    }
    __syncthreads();
    if (threadIdx.x < N_GRAPHS) {
        int g = threadIdx.x;
        float mx = -1e30f;
        for (int c = 0; c < N_CLASSES; ++c) mx = fmaxf(mx, logits[g * N_CLASSES + c]);
        float se = 0.0f;
        for (int c = 0; c < N_CLASSES; ++c) se += expf(logits[g * N_CLASSES + c] - mx);
        float lse = mx + logf(se);
        for (int c = 0; c < N_CLASSES; ++c)
            out[g * N_CLASSES + c] = logits[g * N_CLASSES + c] - lse;
    }
}

// ---------------- workspace layout (bytes, 16B-aligned) ----------------
static const size_t OFF_H     = 0;                          // 10000*512*4 = 20,480,000
static const size_t OFF_AGGR  = 20480000;                   // 10000*512*4
static const size_t OFF_Z16   = 40960000;                   // 10000*512*2
static const size_t OFF_MID16 = 51200000;                   // 10000*512*2
static const size_t OFF_W     = 61440000;                   // bf16 transposed weights
static const size_t SZ_W1L1   = (size_t)HIDDEN * F_IN * 2;  // 131072
static const size_t SZ_W      = (size_t)HIDDEN * HIDDEN * 2;// 524288
static const size_t OFF_SUMS  = OFF_W + SZ_W1L1 + 7 * SZ_W; // 65,241,088
static const size_t OFF_CNT   = OFF_SUMS + (size_t)N_GRAPHS * HIDDEN * 4;

extern "C" void kernel_launch(void* const* d_in, const int* in_sizes, int n_in,
                              void* d_out, int out_size, void* d_ws, size_t ws_size,
                              hipStream_t stream) {
    (void)in_sizes; (void)n_in; (void)out_size; (void)ws_size;

    const float* x    = (const float*)d_in[0];
    const int*   edge = (const int*)d_in[1];
    const int*   batch= (const int*)d_in[2];
    const float* c1w1 = (const float*)d_in[3];
    const float* c1b1 = (const float*)d_in[4];
    const float* c1w2 = (const float*)d_in[5];
    const float* c1b2 = (const float*)d_in[6];
    const float* cw1  = (const float*)d_in[7];   // [3][512][512]
    const float* cb1  = (const float*)d_in[8];   // [3][512]
    const float* cw2  = (const float*)d_in[9];
    const float* cb2  = (const float*)d_in[10];
    const float* fcw  = (const float*)d_in[11];
    const float* fcb  = (const float*)d_in[12];
    float* out = (float*)d_out;

    char* ws = (char*)d_ws;
    float*          h      = (float*)(ws + OFF_H);
    float*          aggr   = (float*)(ws + OFF_AGGR);
    unsigned short* z16    = (unsigned short*)(ws + OFF_Z16);
    unsigned short* mid16  = (unsigned short*)(ws + OFF_MID16);
    unsigned short* w1t_l1 = (unsigned short*)(ws + OFF_W);
    unsigned short* w2t_l1 = (unsigned short*)(ws + OFF_W + SZ_W1L1);
    unsigned short* wt_cv1 = (unsigned short*)(ws + OFF_W + SZ_W1L1 + SZ_W);          // [3]
    unsigned short* wt_cv2 = (unsigned short*)(ws + OFF_W + SZ_W1L1 + 4 * SZ_W);      // [3]
    float*          sums   = (float*)(ws + OFF_SUMS);
    float*          counts = (float*)(ws + OFF_CNT);

    const int* src = edge;
    const int* dst = edge + N_EDGES;

    // ---- convert & transpose weights to bf16 ----
    gin_wt_convert<<<(F_IN * HIDDEN + 255) / 256, 256, 0, stream>>>(c1w1, w1t_l1, F_IN, HIDDEN);
    gin_wt_convert<<<(HIDDEN * HIDDEN + 255) / 256, 256, 0, stream>>>(c1w2, w2t_l1, HIDDEN, HIDDEN);
    for (int i = 0; i < 3; ++i) {
        gin_wt_convert<<<(HIDDEN * HIDDEN + 255) / 256, 256, 0, stream>>>(
            cw1 + (size_t)i * HIDDEN * HIDDEN, wt_cv1 + (size_t)i * HIDDEN * HIDDEN, HIDDEN, HIDDEN);
        gin_wt_convert<<<(HIDDEN * HIDDEN + 255) / 256, 256, 0, stream>>>(
            cw2 + (size_t)i * HIDDEN * HIDDEN, wt_cv2 + (size_t)i * HIDDEN * HIDDEN, HIDDEN, HIDDEN);
    }

    dim3 gemm_grid((N_NODES + BM - 1) / BM, HIDDEN / BN);

    auto run_layer = [&](const float* hin, int F, int logF,
                         const unsigned short* w1t, const float* b1,
                         const unsigned short* w2t, const float* b2) {
        int nf = N_NODES * F;
        gin_fill_zero<<<(nf + 255) / 256, 256, 0, stream>>>(aggr, nf);
        int logF4 = logF - 2;
        int total = N_EDGES << logF4;
        gin_aggregate<<<(total + 255) / 256, 256, 0, stream>>>(
            (const float4*)hin, src, dst, aggr, logF4, total);
        gin_add_convert<<<(nf + 255) / 256, 256, 0, stream>>>(hin, aggr, z16, nf);
        gin_gemm_bias_relu<true ><<<gemm_grid, 256, 0, stream>>>(z16,  w1t, b1, mid16, N_NODES, F,      HIDDEN);
        gin_gemm_bias_relu<false><<<gemm_grid, 256, 0, stream>>>(mid16, w2t, b2, h,     N_NODES, HIDDEN, HIDDEN);
    };

    // layer 1 (input x, F=128), layers 2..4 (h, F=512)
    run_layer(x, F_IN, 7, w1t_l1, c1b1, w2t_l1, c1b2);
    for (int i = 0; i < 3; ++i)
        run_layer(h, HIDDEN, 9,
                  wt_cv1 + (size_t)i * HIDDEN * HIDDEN, cb1 + (size_t)i * HIDDEN,
                  wt_cv2 + (size_t)i * HIDDEN * HIDDEN, cb2 + (size_t)i * HIDDEN);

    // ---- global mean pool + FC + log_softmax ----
    gin_fill_zero<<<(N_GRAPHS * HIDDEN + 255) / 256, 256, 0, stream>>>(sums, N_GRAPHS * HIDDEN);
    gin_fill_zero<<<1, 64, 0, stream>>>(counts, N_GRAPHS);
    int ptotal = N_NODES * (HIDDEN / 4);
    gin_pool_sum<<<(ptotal + 255) / 256, 256, 0, stream>>>((const float4*)h, batch, sums, ptotal);
    gin_count_nodes<<<(N_NODES + 255) / 256, 256, 0, stream>>>(batch, counts);
    gin_fc_logsoftmax<<<1, 256, 0, stream>>>(sums, counts, fcw, fcb, out);
}
